// VectorizedGRU_17489106829869
// MI455X (gfx1250) — matrix operand order repaired and checked by reading, not compiled, not run on hardware
//
#include <hip/hip_runtime.h>
#include <math.h>

// ---------------------------------------------------------------------------
// GRU for MI455X (gfx1250, wave32, WMMA bf16 16x16x32 + TDM prefetch)
// B=32, L=1024, D_IN=512, H=512
// ---------------------------------------------------------------------------

typedef __attribute__((ext_vector_type(16))) __bf16 v16bf;
typedef __attribute__((ext_vector_type(8)))  float  v8f;
typedef __attribute__((ext_vector_type(4)))  unsigned int u32x4;
typedef __attribute__((ext_vector_type(8)))  int i32x8;
typedef __attribute__((ext_vector_type(4)))  int i32x4;

#define Bsz   32
#define Lseq  1024
#define Din   512
#define Hdim  512
#define NG    1536            // 3H: [r(512) | z(512) | c(512)] columns of x_proj
#define KT_W  16              // K tiles (512 / 32)

__device__ __forceinline__ v8f zero8() {
    v8f v;
#pragma unroll
    for (int i = 0; i < 8; ++i) v[i] = 0.0f;
    return v;
}

__device__ __forceinline__ v8f wmma_bf16(v16bf a, v16bf b, v8f c) {
    return __builtin_amdgcn_wmma_f32_16x16x32_bf16(
        false, a, false, b, (short)0, c, false, false);
}

// A-fragment K-offset for (lane, elem): K = kt*32 + (e&7) + (e>=8?16:0) + (lane>=16?8:0)
// B-fragment layout (per lane 16 contiguous bf16): n = nt*16 + (lane&15),
//   k = kt*32 + (e&7) + (e>=8?16:0) + (lane>=16?8:0)   (same k order as A)

// ---------------------------------------------------------------------------
// Pack a weight matrix Wsrc (N rows x ld cols, fp32) into bf16 B-fragments of
// B(k,n) = Wsrc[n, coloff + k].  Per-lane chunk = 16 contiguous bf16 (32 B).
// ---------------------------------------------------------------------------
__global__ void gru_pack_b_kernel(const float* __restrict__ src, int ld, int coloff,
                                  int ntiles, unsigned short* __restrict__ dst_u16,
                                  int nt_total, int nt_base) {
    __bf16* dst = (__bf16*)dst_u16;
    int id = blockIdx.x * blockDim.x + threadIdx.x;
    int total = KT_W * ntiles * 32;
    if (id >= total) return;
    int lane = id & 31;
    int tile = id >> 5;
    int kt = tile / ntiles;
    int nt = tile - kt * ntiles;
    int n  = nt * 16 + (lane & 15);
    int kb = kt * 32 + ((lane >= 16) ? 8 : 0);
    const float* s = src + (size_t)n * ld + coloff + kb;
    float fa[16];
    *(float4*)(fa + 0)  = *(const float4*)(s + 0);
    *(float4*)(fa + 4)  = *(const float4*)(s + 4);
    *(float4*)(fa + 8)  = *(const float4*)(s + 16);
    *(float4*)(fa + 12) = *(const float4*)(s + 20);
    __bf16* d = dst + ((size_t)(kt * nt_total + nt_base + nt) * 32 + lane) * 16;
#pragma unroll
    for (int e = 0; e < 16; ++e) d[e] = (__bf16)fa[e];
}

// ---------------------------------------------------------------------------
// x_proj[row, g] = sum_k x[row,k] * Wx(g,k) + bias(g)   (bf16 out)
// One wave computes a 16 x 64 strip (4 N-tiles, shared A fragment).
// Grid: 2048 M-tiles * 24 N-groups = 49152 waves = 6144 blocks of 256.
// ---------------------------------------------------------------------------
__global__ __launch_bounds__(256)
void gru_xproj_gemm_kernel(const float* __restrict__ x,
                           const unsigned short* __restrict__ wx_u16,
                           const float* __restrict__ b_ru,
                           const float* __restrict__ b_c,
                           unsigned short* __restrict__ xproj_u16) {
    const __bf16* WxP = (const __bf16*)wx_u16;
    __bf16* xproj = (__bf16*)xproj_u16;
    int wave = blockIdx.x * 8 + (threadIdx.x >> 5);
    int lane = threadIdx.x & 31;
    int mt = wave / 24;
    int ng = wave - mt * 24;
    int nt0 = ng * 4;

    int rowA  = mt * 16 + (lane & 15);
    int koff  = (lane >= 16) ? 8 : 0;

    v8f acc[4];
#pragma unroll
    for (int j = 0; j < 4; ++j) acc[j] = zero8();

    for (int kt = 0; kt < KT_W; ++kt) {
        const float* xr = x + (size_t)rowA * Din + kt * 32 + koff;
        float fa[16];
        *(float4*)(fa + 0)  = *(const float4*)(xr + 0);
        *(float4*)(fa + 4)  = *(const float4*)(xr + 4);
        *(float4*)(fa + 8)  = *(const float4*)(xr + 16);
        *(float4*)(fa + 12) = *(const float4*)(xr + 20);
        v16bf a;
#pragma unroll
        for (int e = 0; e < 16; ++e) a[e] = (__bf16)fa[e];
#pragma unroll
        for (int j = 0; j < 4; ++j) {
            v16bf b = *(const v16bf*)(WxP +
                ((size_t)(kt * 96 + nt0 + j) * 32 + lane) * 16);
            acc[j] = wmma_bf16(a, b, acc[j]);
        }
    }

    int colL = lane & 15;
    int mhi  = (lane >= 16) ? 8 : 0;
#pragma unroll
    for (int j = 0; j < 4; ++j) {
        int n = (nt0 + j) * 16 + colL;
        float bias = (n < 1024) ? b_ru[n] : b_c[n - 1024];
#pragma unroll
        for (int r = 0; r < 8; ++r) {
            int row = mt * 16 + r + mhi;
            xproj[(size_t)row * NG + n] = (__bf16)(acc[j][r] + bias);
        }
    }
}

// ---------------------------------------------------------------------------
// TDM: prefetch the step-t x_proj slice (16 rows x 1536 bf16, row stride
// L*1536 elements) into LDS.  2-D descriptor, groups 2/3 unused (zero).
// Issued by one wave; completion via TENSORcnt + workgroup barrier.
// This toolchain's builtin takes 6 args (g0,g1,g2,g3,g4,cpol).
// ---------------------------------------------------------------------------
__device__ __forceinline__ void tdm_load_slice(const unsigned short* xprojB,
                                               unsigned lds_off, int bb, int t) {
#if __has_builtin(__builtin_amdgcn_tensor_load_to_lds)
    unsigned long long g = (unsigned long long)(uintptr_t)xprojB +
        ((unsigned long long)((size_t)bb * Lseq + t) * NG) * 2ull;
    const unsigned long long stride0 = (unsigned long long)Lseq * NG; // elements
    u32x4 g0;
    g0[0] = 1u;                                                // count=1, no gather
    g0[1] = lds_off;                                           // lds_addr
    g0[2] = (unsigned)(g & 0xFFFFFFFFull);                     // global_addr lo
    g0[3] = (unsigned)((g >> 32) & 0x01FFFFFFull) | (2u << 30); // addr hi | type=2
    i32x8 g1;
    g1[0] = (1 << 16);                          // workgroup_mask=0, data_size=1 (2B)
    g1[1] = (int)((NG & 0xFFFF) << 16);         // tensor_dim0[15:0]
    g1[2] = (NG >> 16) | (16 << 16);            // tensor_dim0[31:16] | tensor_dim1 lo
    g1[3] = (int)(NG << 16);                    // tensor_dim1 hi | tile_dim0=1536
    g1[4] = 16;                                 // tile_dim1=16, tile_dim2=0
    g1[5] = (int)(stride0 & 0xFFFFFFFFull);     // tensor_dim0_stride lo
    g1[6] = (int)(stride0 >> 32);               // stride hi | dim1_stride lo = 0
    g1[7] = 0;
    i32x4 z4;
    z4[0] = 0; z4[1] = 0; z4[2] = 0; z4[3] = 0;
    i32x8 z8;
#pragma unroll
    for (int i = 0; i < 8; ++i) z8[i] = 0;
    __builtin_amdgcn_tensor_load_to_lds(g0, g1, z4, z4, z8, 0);
#endif
}

// ---------------------------------------------------------------------------
// Recurrent scan.  2 blocks x 512 threads (16 waves); block owns 16 batch rows.
// h kept in LDS as bf16 A-fragments; x_proj slices DMA'd in by the TDM,
// double-buffered one step ahead.  Per step:
//   gates = sigmoid(h @ Wh_ru^T + xg)  -> r written as (r*h) fragments, z in regs
//   cand  = tanh((r*h) @ Wh_c^T + xc)
//   h_new = (1-z)*h + z*cand  -> out + LDS fragment update
// Wave w gate tiles: {2w,2w+1, 32+2w,33+2w}; cand tiles {2w,2w+1}  (z cols match).
// ---------------------------------------------------------------------------
__global__ __launch_bounds__(512)
void gru_scan_kernel(const float* __restrict__ init_h,
                     const unsigned short* __restrict__ xprojB,
                     const unsigned short* __restrict__ whru_u16,
                     const unsigned short* __restrict__ whc_u16,
                     float* __restrict__ out) {
    const __bf16* WhruP = (const __bf16*)whru_u16;
    const __bf16* WhcP  = (const __bf16*)whc_u16;

    __shared__ __align__(32) __bf16 h_frag[16][32][16];   // 16 KB
    __shared__ __align__(32) __bf16 rh_frag[16][32][16];  // 16 KB
    __shared__ __align__(32) __bf16 xgbuf[2][16][NG];     // 96 KB (double buffer)

    const int tid  = threadIdx.x;
    const int wave = tid >> 5;
    const int lane = tid & 31;
    const int bb   = blockIdx.x * 16;          // first batch row of this block
    const int Ml   = lane & 15;
    const int khalf = (lane >= 16) ? 8 : 0;
    const int mhi   = (lane >= 16) ? 8 : 0;

    // init h fragments from init_states
    {
        int kt = wave;
#pragma unroll
        for (int e = 0; e < 16; ++e) {
            int col = kt * 32 + (e & 7) + ((e >= 8) ? 16 : 0) + khalf;
            h_frag[kt][lane][e] = (__bf16)init_h[(size_t)(bb + Ml) * Hdim + col];
        }
    }

    // prologue: DMA slice for t=0 into buffer 0
    if (wave == 0) {
        tdm_load_slice(xprojB, (unsigned)(uintptr_t)&xgbuf[0][0][0], bb, 0);
    }

    const int ntg[4] = {2 * wave, 2 * wave + 1, 32 + 2 * wave, 33 + 2 * wave};
    float zreg[2][8];

    for (int t = 0; t < Lseq; ++t) {
        // prefetch next slice; publish current one
        if (wave == 0) {
            if (t + 1 < Lseq) {
                tdm_load_slice(xprojB,
                               (unsigned)(uintptr_t)&xgbuf[(t + 1) & 1][0][0],
                               bb, t + 1);
#if __has_builtin(__builtin_amdgcn_s_wait_tensorcnt)
                __builtin_amdgcn_s_wait_tensorcnt(1);   // slice t resident
#endif
            } else {
#if __has_builtin(__builtin_amdgcn_s_wait_tensorcnt)
                __builtin_amdgcn_s_wait_tensorcnt(0);
#endif
            }
        }
        __syncthreads();
        const __bf16* xg = &xgbuf[t & 1][0][0];   // [M][col], col in [0,1536)

        // ---------------- gates GEMM: 16x1024 over this block ----------------
        v8f acc[4];
#pragma unroll
        for (int j = 0; j < 4; ++j) acc[j] = zero8();

        for (int kt = 0; kt < KT_W; ++kt) {
            v16bf a = *(const v16bf*)&h_frag[kt][lane][0];
#pragma unroll
            for (int j = 0; j < 4; ++j) {
                v16bf b = *(const v16bf*)(WhruP +
                    ((size_t)(kt * 64 + ntg[j]) * 32 + lane) * 16);
                acc[j] = wmma_bf16(a, b, acc[j]);
            }
        }

#pragma unroll
        for (int j = 0; j < 4; ++j) {
            int n = ntg[j] * 16 + Ml;
#pragma unroll
            for (int r = 0; r < 8; ++r) {
                int M = r + mhi;
                float pre = acc[j][r] + (float)xg[M * NG + n];
                float g = 1.0f / (1.0f + __expf(-pre));
                if (j < 2) {
                    // r gate: store (r*h) as A fragment for cand GEMM
                    int kol   = n & 31;
                    int lane2 = M + (((kol >> 3) & 1) ? 16 : 0);
                    int e2    = (kol & 7) + ((kol >= 16) ? 8 : 0);
                    float hold = (float)h_frag[n >> 5][lane2][e2];
                    rh_frag[n >> 5][lane2][e2] = (__bf16)(g * hold);
                } else {
                    zreg[j - 2][r] = g;     // z cols == this wave's cand cols
                }
            }
        }
        __syncthreads();

        // ---------------- cand GEMM: 16x512, then h update ----------------
        v8f cacc[2];
#pragma unroll
        for (int j = 0; j < 2; ++j) cacc[j] = zero8();

        for (int kt = 0; kt < KT_W; ++kt) {
            v16bf a = *(const v16bf*)&rh_frag[kt][lane][0];
#pragma unroll
            for (int j = 0; j < 2; ++j) {
                v16bf b = *(const v16bf*)(WhcP +
                    ((size_t)(kt * 32 + 2 * wave + j) * 32 + lane) * 16);
                cacc[j] = wmma_bf16(a, b, cacc[j]);
            }
        }

#pragma unroll
        for (int j = 0; j < 2; ++j) {
            int c   = (2 * wave + j) * 16 + Ml;   // c>>5 == wave
            int kol = c & 31;
#pragma unroll
            for (int r = 0; r < 8; ++r) {
                int M = r + mhi;
                int brow = bb + M;
                float pre = cacc[j][r] + (float)xg[M * NG + 1024 + c];
                float cand = tanhf(pre);
                int lane2 = M + (((kol >> 3) & 1) ? 16 : 0);
                int e2    = (kol & 7) + ((kol >= 16) ? 8 : 0);
                float hold = (float)h_frag[wave][lane2][e2];
                float z = zreg[j][r];
                float hn = (1.0f - z) * hold + z * cand;
                out[(size_t)(brow * Lseq + t) * Hdim + c] = hn;
                h_frag[wave][lane2][e2] = (__bf16)hn;   // own (lane2,e2) slot
            }
        }
        __syncthreads();
    }

    // ---------------- h_last ----------------
    {
        int kt = wave;
        size_t base = (size_t)Bsz * Lseq * Hdim;
#pragma unroll
        for (int e = 0; e < 16; ++e) {
            int col = kt * 32 + (e & 7) + ((e >= 8) ? 16 : 0) + khalf;
            out[base + (size_t)(bb + Ml) * Hdim + col] = (float)h_frag[kt][lane][e];
        }
    }
}

// ---------------------------------------------------------------------------
// Host launcher
// ---------------------------------------------------------------------------
extern "C" void kernel_launch(void* const* d_in, const int* in_sizes, int n_in,
                              void* d_out, int out_size, void* d_ws, size_t ws_size,
                              hipStream_t stream) {
    const float* x     = (const float*)d_in[0];  // (32,1024,512)
    const float* inith = (const float*)d_in[1];  // (32,512)
    const float* W_ru  = (const float*)d_in[2];  // (1024, 1024) hidden cols [0,512), x cols [512,1024)
    const float* b_ru  = (const float*)d_in[3];  // (1024)
    const float* W_c   = (const float*)d_in[4];  // (512, 1024)
    const float* b_c   = (const float*)d_in[5];  // (512)
    float* out = (float*)d_out;

    char* ws = (char*)d_ws;
    const size_t sz_wx   = (size_t)Din * NG * 2;        // 1.5 MB bf16
    const size_t sz_whru = (size_t)Hdim * 1024 * 2;     // 1.0 MB
    const size_t sz_whc  = (size_t)Hdim * Hdim * 2;     // 0.5 MB
    unsigned short* WxP   = (unsigned short*)(ws);
    unsigned short* WhruP = (unsigned short*)(ws + sz_wx);
    unsigned short* WhcP  = (unsigned short*)(ws + sz_wx + sz_whru);
    unsigned short* xproj = (unsigned short*)(ws + sz_wx + sz_whru + sz_whc); // 96 MB bf16

    // Pack weights into bf16 B-fragments.
    // x_proj weights: B(k,g) = Wx(g,k); g<1024 from W_ru[:,512:], g>=1024 from W_c[:,512:]
    gru_pack_b_kernel<<<128, 256, 0, stream>>>(W_ru, 1024, 512, 64, WxP, 96, 0);
    gru_pack_b_kernel<<<64,  256, 0, stream>>>(W_c,  1024, 512, 32, WxP, 96, 64);
    // recurrent weights: B(k,n) = Wh(n,k) from W_ru[:,:512] and W_c[:,:512]
    gru_pack_b_kernel<<<128, 256, 0, stream>>>(W_ru, 1024, 0, 64, WhruP, 64, 0);
    gru_pack_b_kernel<<<64,  256, 0, stream>>>(W_c,  1024, 0, 32, WhcP,  32, 0);

    // Input projection: 2048 M-tiles x 24 N-groups -> 6144 blocks of 256.
    gru_xproj_gemm_kernel<<<6144, 256, 0, stream>>>(x, WxP, b_ru, b_c, xproj);

    // Recurrent scan: 2 persistent workgroups (16 batch rows each).
    gru_scan_kernel<<<2, 512, 0, stream>>>(inith, xproj, WhruP, WhcP, out);
}
